// PhonemeVisemeDualStatePlanner_28338194219145
// MI455X (gfx1250) — compile-verified, matrix operand order checked by reading, not run on hardware
//
#include <hip/hip_runtime.h>
#include <math.h>

// ---------------------------------------------------------------------------
// MI455X / gfx1250 implementation. Compute-bound forward pass; all dense
// layers routed through V_WMMA_F32_16X16X32_BF16 (fp32 accumulate), with
// block tiles staged into LDS via GLOBAL_LOAD_ASYNC_TO_LDS_B128 (ASYNCcnt)
// when the toolchain exposes the builtin.
// ---------------------------------------------------------------------------

typedef __bf16 bf16_t;
typedef __bf16 bf16x8  __attribute__((ext_vector_type(8)));
typedef __bf16 bf16x16 __attribute__((ext_vector_type(16)));
typedef float  f32x8   __attribute__((ext_vector_type(8)));
typedef int    v4i     __attribute__((vector_size(16)));   // matches builtin prototype

#ifndef __has_builtin
#define __has_builtin(x) 0
#endif

#if __has_builtin(__builtin_amdgcn_global_load_async_to_lds_b128)
#define ASYNC_LDS 1
#else
#define ASYNC_LDS 0
#endif

#define AS1 __attribute__((address_space(1)))
#define AS3 __attribute__((address_space(3)))

// stage 16 bytes global -> LDS (async DMA path if available)
__device__ __forceinline__ void stage16(bf16_t* l, const bf16_t* g) {
#if ASYNC_LDS
  __builtin_amdgcn_global_load_async_to_lds_b128(
      (AS1 v4i*)(__UINTPTR_TYPE__)g, (AS3 v4i*)l, 0, 0);
#else
  *(bf16x8*)l = *(const bf16x8*)g;   // load + ds_store fallback
#endif
}
__device__ __forceinline__ void stage_wait() {
#if ASYNC_LDS
#if __has_builtin(__builtin_amdgcn_s_wait_asynccnt)
  __builtin_amdgcn_s_wait_asynccnt(0);
#else
  asm volatile("s_wait_asynccnt 0x0" ::: "memory");
#endif
#endif
  __syncthreads();
}

// ---------------------------------------------------------------------------
// Generic batched GEMM:  C[b][M,N] (+)= act( A[b][M,K]_bf16 * Wt[b][N,K]_bf16 + bias )
// Wt is stored row-major [N,K] (B transposed) so both operands are contiguous
// along K. Block = 256 threads = 8 wave32s; wave grid 2(M)x4(N); each wave a
// 32x32 tile = 2x2 WMMA 16x16 tiles. K panel of 64 staged in LDS per round
// (sA 8KB + sB 16KB), 2 unrolled K=32 WMMA sub-steps -> 8 v_wmma per barrier.
// Fragment layout per CDNA5 ISA (16-bit A 16x32): lane<16 -> row=lane,
// K {g*8..g*8+7} in elems 0..7 and {16+g*8..} in elems 8..15 with g=lane>>4.
// ---------------------------------------------------------------------------
__global__ __launch_bounds__(256) void k_gemm(
    const bf16_t* __restrict__ A,  long long aS1, long long aS2, int lda,
    const bf16_t* __restrict__ Bt, long long bS1, long long bS2, int ldb,
    float* __restrict__ C,         long long cS1, long long cS2, int ldc,
    const float* __restrict__ bias, int M, int N, int K,
    int inner, int act, int resAdd)
{
  __shared__ bf16_t sA[64 * 64];     // [row][k] 64x64 bf16 = 8KB
  __shared__ bf16_t sB[128 * 64];    // [col][k] 128x64 bf16 = 16KB

  int z  = blockIdx.z;
  int b1 = z / inner, b2 = z % inner;
  A  += b1 * aS1 + b2 * aS2;
  Bt += b1 * bS1 + b2 * bS2;
  C  += b1 * cS1 + b2 * cS2;

  int tid  = threadIdx.x;
  int lane = tid & 31;
  int w    = tid >> 5;
  int wm   = w & 1;          // 0..1 over M
  int wn   = w >> 1;         // 0..3 over N
  int l15  = lane & 15;
  int g    = lane >> 4;      // lane group for K split
  int rowBase = blockIdx.y * 64;
  int colBase = blockIdx.x * 128;
  int wrow = wm * 32;
  int wcol = wn * 32;

  // per-thread staging slots: A = 512 chunks (2/thread), B = 1024 (4/thread)
  const bf16_t* aG[2]; bf16_t* aL[2];
#pragma unroll
  for (int i = 0; i < 2; ++i) {
    int c = tid + i * 256;
    int r = c >> 3, sub = c & 7;
    int gr = rowBase + r; if (gr > M - 1) gr = M - 1;
    aG[i] = A + (long long)gr * lda + sub * 8;
    aL[i] = sA + r * 64 + sub * 8;
  }
  const bf16_t* bG[4]; bf16_t* bL[4];
#pragma unroll
  for (int i = 0; i < 4; ++i) {
    int c = tid + i * 256;
    int r = c >> 3, sub = c & 7;
    int gc = colBase + r; if (gc > N - 1) gc = N - 1;
    bG[i] = Bt + (long long)gc * ldb + sub * 8;
    bL[i] = sB + r * 64 + sub * 8;
  }

  f32x8 acc[2][2] = {};

  for (int kb = 0; kb < K; kb += 64) {
#pragma unroll
    for (int i = 0; i < 2; ++i) stage16(aL[i], aG[i] + kb);
#pragma unroll
    for (int i = 0; i < 4; ++i) stage16(bL[i], bG[i] + kb);
    if (kb + 64 < K) {  // global_prefetch_b8 of next K panel
      __builtin_prefetch(aG[0] + kb + 64, 0, 1);
      __builtin_prefetch(bG[0] + kb + 64, 0, 1);
    }
    stage_wait();        // s_wait_asynccnt 0 + barrier

#pragma unroll
    for (int kk = 0; kk < 64; kk += 32) {
      bf16x16 af[2], bfm[2];
#pragma unroll
      for (int i = 0; i < 2; ++i) {
        const bf16_t* p = sA + (wrow + i * 16 + l15) * 64 + kk + g * 8;
        bf16x8 lo = *(const bf16x8*)(p);
        bf16x8 hi = *(const bf16x8*)(p + 16);
        af[i] = __builtin_shufflevector(lo, hi, 0,1,2,3,4,5,6,7,8,9,10,11,12,13,14,15);
      }
#pragma unroll
      for (int j = 0; j < 2; ++j) {
        const bf16_t* p = sB + (wcol + j * 16 + l15) * 64 + kk + g * 8;
        bf16x8 lo = *(const bf16x8*)(p);
        bf16x8 hi = *(const bf16x8*)(p + 16);
        bfm[j] = __builtin_shufflevector(lo, hi, 0,1,2,3,4,5,6,7,8,9,10,11,12,13,14,15);
      }
#pragma unroll
      for (int i = 0; i < 2; ++i)
#pragma unroll
        for (int j = 0; j < 2; ++j)
          acc[i][j] = __builtin_amdgcn_wmma_f32_16x16x32_bf16(
              false, af[i], false, bfm[j], (short)0, acc[i][j], false, false);
    }
    __syncthreads();     // protect LDS before next staging round
  }

  // C/D layout: lanes 0-15 -> M = e (VGPR e), lanes 16-31 -> M = 8+e; N = l15.
#pragma unroll
  for (int i = 0; i < 2; ++i) {
#pragma unroll
    for (int j = 0; j < 2; ++j) {
      int col = colBase + wcol + j * 16 + l15;
      if (col >= N) continue;
      float bv = bias ? bias[col] : 0.0f;
#pragma unroll
      for (int e = 0; e < 8; ++e) {
        int row = rowBase + wrow + i * 16 + g * 8 + e;
        if (row >= M) continue;
        float v = acc[i][j][e] + bv;
        if (act == 1)      v = 0.5f * v * (1.0f + erff(v * 0.70710678118654752f)); // exact gelu
        else if (act == 2) v = 1.0f / (1.0f + __expf(-v));                          // sigmoid
        long long idx = (long long)row * ldc + col;
        if (resAdd) C[idx] += v; else C[idx] = v;
      }
    }
  }
}

// -------------------------- small device helpers ---------------------------
__device__ __forceinline__ float blkSum(float v, float* buf) {
  int t = threadIdx.x;
  buf[t] = v; __syncthreads();
  for (int s = 128; s > 0; s >>= 1) { if (t < s) buf[t] += buf[t + s]; __syncthreads(); }
  float r = buf[0]; __syncthreads();
  return r;
}
__device__ __forceinline__ float blkMax(float v, float* buf) {
  int t = threadIdx.x;
  buf[t] = v; __syncthreads();
  for (int s = 128; s > 0; s >>= 1) { if (t < s) buf[t] = fmaxf(buf[t], buf[t + s]); __syncthreads(); }
  float r = buf[0]; __syncthreads();
  return r;
}

// -------------------------- elementwise kernels ----------------------------
__global__ void k_cvt_bf16(bf16_t* __restrict__ dst, const float* __restrict__ src, long long n) {
  long long i = (long long)blockIdx.x * 256 + threadIdx.x;
  if (i < n) dst[i] = (bf16_t)src[i];
}

// transpose+convert: dst[z][c*R + r] = src[b1*sS1 + b2*sS2 + r*rowStride + c]
__global__ void k_transpose_bf16(bf16_t* __restrict__ dst, const float* __restrict__ src,
                                 int R, int Cn, int rowStride,
                                 long long sS1, long long sS2, int inner, long long dStride) {
  int z = blockIdx.z; int b1 = z / inner, b2 = z % inner;
  src += (long long)b1 * sS1 + (long long)b2 * sS2;
  dst += (long long)z * dStride;
  long long i = (long long)blockIdx.x * 256 + threadIdx.x;
  if (i < (long long)R * Cn) {
    int r = (int)(i / Cn), c = (int)(i % Cn);
    dst[(long long)c * R + r] = (bf16_t)src[(long long)r * rowStride + c];
  }
}

__global__ __launch_bounds__(256) void k_layernorm(
    const float* __restrict__ x, const float* __restrict__ gw, const float* __restrict__ bw,
    float* __restrict__ y, bf16_t* __restrict__ ybf, int D) {
  __shared__ float red[256];
  long long row = blockIdx.x;
  const float* xr = x + row * D;
  int t = threadIdx.x;
  float s = 0.f;
  for (int i = t; i < D; i += 256) s += xr[i];
  float mean = blkSum(s, red) / (float)D;
  float v = 0.f;
  for (int i = t; i < D; i += 256) { float d = xr[i] - mean; v += d * d; }
  float rstd = rsqrtf(blkSum(v, red) / (float)D + 1e-5f);
  for (int i = t; i < D; i += 256) {
    float o = (xr[i] - mean) * rstd * gw[i] + bw[i];
    if (y)   y[row * D + i]   = o;
    if (ybf) ybf[row * D + i] = (bf16_t)o;
  }
}

// self-attn softmax over S=512 with key padding mask; grid (S, NH, B)
__global__ __launch_bounds__(256) void k_softmax_self(
    float* __restrict__ sc, bf16_t* __restrict__ scbf, const int* __restrict__ vlens,
    int Sn, float scale) {
  __shared__ float red[256];
  int q = blockIdx.x, h = blockIdx.y, b = blockIdx.z;
  long long base = ((long long)(b * gridDim.y + h) * Sn + q) * Sn;
  float* row = sc + base;
  int len = vlens[b];
  int t = threadIdx.x;
  float vals[2]; float mx = -3.4e38f;
#pragma unroll
  for (int i = 0; i < 2; ++i) {
    int c = t + i * 256;
    float v = (c < len) ? row[c] * scale : -3.4e38f;
    vals[i] = v; mx = fmaxf(mx, v);
  }
  float m = blkMax(mx, red);
  float sum = 0.f;
#pragma unroll
  for (int i = 0; i < 2; ++i) {
    int c = t + i * 256;
    float e = (c < len) ? __expf(vals[i] - m) : 0.f;
    vals[i] = e; sum += e;
  }
  float Z = blkSum(sum, red);
  float inv = 1.0f / Z;
#pragma unroll
  for (int i = 0; i < 2; ++i) {
    int c = t + i * 256;
    float p = vals[i] * inv;
    row[c] = p; scbf[base + c] = (bf16_t)p;
  }
}

// fused cross-attention post-processing; one block per (s,b); blockDim == T == 256
__global__ __launch_bounds__(256) void k_cross(
    float* __restrict__ logits, const float* __restrict__ vlog,
    const int* __restrict__ tok, const int* __restrict__ vtab,
    const int* __restrict__ tlens,
    float* __restrict__ exp_pos, float* __restrict__ lw, bf16_t* __restrict__ lwbf,
    float* __restrict__ entropy, float* __restrict__ compat_acc,
    int Sn, int Tn, int Cn, int Vn, float scale, float cscale, float inv2r2, int accumulate) {
  __shared__ float red[256];
  __shared__ float vlp[32];
  __shared__ float sh[1];
  int s = blockIdx.x, b = blockIdx.y;
  int t = threadIdx.x;
  const float NEGINF = -__builtin_inff();

  long long vbase = ((long long)b * Sn + s) * Cn;
  if (t == 0) {
    float mx = -3.4e38f;
    for (int c = 0; c < Cn; ++c) mx = fmaxf(mx, vlog[vbase + c]);
    float sm = 0.f;
    for (int c = 0; c < Cn; ++c) sm += __expf(vlog[vbase + c] - mx);
    float lse = mx + __logf(sm);
    for (int c = 0; c < Cn; ++c) vlp[c] = vlog[vbase + c] - lse;
  }
  __syncthreads();

  int tlen = tlens[b];
  bool pad = (t >= tlen);
  long long idx = ((long long)b * Sn + s) * Tn + t;
  float lg = logits[idx] * scale;
  int tk = tok[b * Tn + t]; tk = tk < 0 ? 0 : (tk > Vn - 1 ? Vn - 1 : tk);
  int gid = vtab[tk];       gid = gid < 0 ? 0 : (gid > Cn - 1 ? Cn - 1 : gid);
  float comp = pad ? 0.f : vlp[gid];
  float lgc = pad ? NEGINF : (lg + cscale * comp);
  logits[idx] = lgc;                      // biased+masked logits (ref output)

  // raw softmax + expected position
  float m = blkMax(lgc, red);
  float e = pad ? 0.f : __expf(lgc - m);
  float Z = blkSum(e, red);
  float raw = e / Z;
  float ep = blkSum(raw * (float)t, red);
  if (t == 0) { exp_pos[(long long)b * Sn + s] = ep; sh[0] = ep; }
  __syncthreads();
  ep = sh[0];

  // windowed softmax
  float d = (float)t - ep;
  float ll = pad ? NEGINF : (lgc - d * d * inv2r2);
  float m2 = blkMax(ll, red);
  float e2 = pad ? 0.f : __expf(ll - m2);
  float Z2 = blkSum(e2, red);
  float w = e2 / Z2;
  lw[idx] = w; lwbf[idx] = (bf16_t)w;

  float wc = fmaxf(w, 1e-8f);
  float ent = blkSum(-wc * __logf(wc), red);
  if (t == 0) entropy[(long long)b * Sn + s] = ent;

  if (accumulate) {
    float cs = blkSum(fabsf(comp), red);
    if (t == 0) atomicAdd(compat_acc, cs);
  }
}

__global__ void k_concat2(float* __restrict__ gi, const float* __restrict__ x,
                          const float* __restrict__ tc, int D2, int D, long long n) {
  long long i = (long long)blockIdx.x * 256 + threadIdx.x;
  if (i < n) {
    long long r = i / D2; int c = (int)(i % D2);
    gi[i] = (c < D) ? x[r * D + c] : tc[r * D + (c - D)];
  }
}

__global__ void k_add_rowvec(float* __restrict__ x, const float* __restrict__ v,
                             int SD, int D, long long n) {
  long long i = (long long)blockIdx.x * 256 + threadIdx.x;
  if (i < n) { long long b = i / SD; x[i] += v[b * D + (i % D)]; }
}

__global__ void k_gate_merge(float* __restrict__ x, const float* __restrict__ tc,
                             const float* __restrict__ gate, int D, long long n) {
  long long i = (long long)blockIdx.x * 256 + threadIdx.x;
  if (i < n) x[i] += gate[i / D] * tc[i];
}

__global__ void k_copy(float* __restrict__ dst, const float* __restrict__ src, long long n) {
  long long i = (long long)blockIdx.x * 256 + threadIdx.x;
  if (i < n) dst[i] = src[i];
}

__global__ void k_setzero(float* p, int n) {
  int i = blockIdx.x * 256 + threadIdx.x;
  if (i < n) p[i] = 0.f;
}

__global__ void k_scale_from(float* dst, const float* src, float scale) { dst[0] = src[0] * scale; }

__global__ __launch_bounds__(256) void k_mono(const float* __restrict__ ep,
                                              const int* __restrict__ vlens,
                                              float* __restrict__ out, int Bn, int Sn) {
  __shared__ float rs[256]; __shared__ float rc[256];
  float s = 0.f, c = 0.f;
  for (int i = threadIdx.x; i < Bn * (Sn - 1); i += 256) {
    int b = i / (Sn - 1), s0 = i % (Sn - 1);
    if (s0 + 1 < vlens[b]) {
      float d = ep[(long long)b * Sn + s0] - ep[(long long)b * Sn + s0 + 1];
      if (d > 0.f) s += d;
      c += 1.f;
    }
  }
  int t = threadIdx.x;
  rs[t] = s; rc[t] = c; __syncthreads();
  for (int st = 128; st > 0; st >>= 1) {
    if (t < st) { rs[t] += rs[t + st]; rc[t] += rc[t + st]; }
    __syncthreads();
  }
  if (t == 0) out[0] = rs[0] / fmaxf(rc[0], 1.f);
}

// ---------------------------------------------------------------------------
extern "C" void kernel_launch(void* const* d_in, const int* in_sizes, int n_in,
                              void* d_out, int out_size, void* d_ws, size_t ws_size,
                              hipStream_t stream) {
  const int B = 8, S = 512, T = 256, D = 1024, HID = 4096, NH = 16, C = 20, V = 32000, HD = 64;
  const int BS = B * S, BT = B * T;
  const int D3 = 3 * D, D2 = 2 * D;

  const float* in_video = (const float*)d_in[0];
  const int*   vlens    = (const int*)d_in[1];
  const float* in_tm    = (const float*)d_in[2];
  const int*   tlens    = (const int*)d_in[3];
  const float* in_tsum  = (const float*)d_in[4];
  const float* in_rsum  = (const float*)d_in[5];
  const int*   in_tok   = (const int*)d_in[6];
  const int*   in_vtab  = (const int*)d_in[7];

  // params: JAX pytree flatten order (dict keys sorted)
  int pi = 8;
  auto P = [&](int i) { return (const float*)d_in[i]; };
  struct DenseP { const float* b; const float* w; };
  struct LnP    { const float* b; const float* g; };
  auto nextDense = [&]() { DenseP d{P(pi), P(pi + 1)}; pi += 2; return d; };
  auto nextLn    = [&]() { LnP l{P(pi), P(pi + 1)}; pi += 2; return l; };
  struct HeadP { DenseP fc1, fc2; LnP ln; };
  auto nextHead = [&]() { HeadP h; h.fc1 = nextDense(); h.fc2 = nextDense(); h.ln = nextLn(); return h; };

  HeadP h_activity = nextHead();
  LnP   context_norm = nextLn();
  LnP   input_norm = nextLn();
  struct LayerP { DenseP attn_in, attn_out, ck, cq, cv, f1, f2; LnP fn; DenseP g1, g2; LnP gn, san; DenseP top, vsp; };
  LayerP L[2];
  for (int l = 0; l < 2; ++l) {
    L[l].attn_in = nextDense(); L[l].attn_out = nextDense();
    L[l].ck = nextDense(); L[l].cq = nextDense(); L[l].cv = nextDense();
    L[l].f1 = nextDense(); L[l].f2 = nextDense();
    L[l].fn = nextLn();
    L[l].g1 = nextDense(); L[l].g2 = nextDense();
    L[l].gn = nextLn(); L[l].san = nextLn();
    L[l].top = nextDense(); L[l].vsp = nextDense();
  }
  LnP   output_norm = nextLn();
  DenseP ref_global = nextDense();
  HeadP h_remaining = nextHead();
  HeadP h_stop = nextHead();
  DenseP text_global = nextDense();
  DenseP text_proj = nextDense();
  DenseP video_proj = nextDense();
  HeadP h_viseme = nextHead();

  // ----- workspace bump allocator -----
  size_t off = 0;
  auto alloc = [&](size_t bytes) -> void* {
    void* p = (char*)d_ws + off;
    off += (bytes + 255) & ~(size_t)255;
    return p;
  };

  float* f_x    = (float*)alloc((size_t)BS * D * 4);
  float* f_a    = (float*)alloc((size_t)BS * D * 4);
  float* f_tctx = (float*)alloc((size_t)BS * D * 4);
  float* f_qkv  = (float*)alloc((size_t)BS * D3 * 4);
  float* f_big  = (float*)alloc((size_t)BS * HID * 4);
  float* f_sc   = (float*)alloc((size_t)B * NH * S * S * 4);
  float* f_tm   = (float*)alloc((size_t)BT * D * 4);
  float* f_k2   = (float*)alloc((size_t)BT * D * 4);
  float* f_v2   = (float*)alloc((size_t)BT * D * 4);
  float* f_lg   = (float*)alloc((size_t)BS * T * 4);
  float* f_lw   = (float*)alloc((size_t)BS * T * 4);
  float* f_vlog = (float*)alloc((size_t)BS * C * 4);
  float* f_gi   = (float*)alloc((size_t)BS * D2 * 4);
  float* f_gate = (float*)alloc((size_t)BS * 4);
  float* f_ep   = (float*)alloc((size_t)BS * 4);
  float* f_ent  = (float*)alloc((size_t)BS * 4);
  float* f_tg   = (float*)alloc((size_t)B * D * 4);
  float* f_rg   = (float*)alloc((size_t)B * D * 4);
  float* f_acc  = (float*)alloc(256);

  bf16_t* b_d0   = (bf16_t*)alloc((size_t)BS * D * 2);
  bf16_t* b_d1   = (bf16_t*)alloc((size_t)BS * D * 2);
  bf16_t* b_qkv  = (bf16_t*)alloc((size_t)BS * D3 * 2);
  bf16_t* b_sc   = (bf16_t*)alloc((size_t)B * NH * S * S * 2);
  bf16_t* b_vt   = (bf16_t*)alloc((size_t)BS * D * 2);
  bf16_t* b_tm   = (bf16_t*)alloc((size_t)BT * D * 2);
  bf16_t* b_tmin = (bf16_t*)alloc((size_t)BT * D * 2);
  bf16_t* b_k2   = (bf16_t*)alloc((size_t)BT * D * 2);
  bf16_t* b_v2t  = (bf16_t*)alloc((size_t)BT * D * 2);
  bf16_t* b_lw   = (bf16_t*)alloc((size_t)BS * T * 2);
  bf16_t* b_big  = (bf16_t*)alloc((size_t)BS * HID * 2);
  bf16_t* b_gi   = (bf16_t*)alloc((size_t)BS * D2 * 2);
  bf16_t* b_ts   = (bf16_t*)alloc((size_t)B * D * 2);
  bf16_t* b_rs   = (bf16_t*)alloc((size_t)B * D * 2);

  // ----- launch helpers -----
  auto cvt = [&](bf16_t* dst, const float* src, long long n) {
    k_cvt_bf16<<<dim3((unsigned)((n + 255) / 256)), 256, 0, stream>>>(dst, src, n);
  };
  auto packT = [&](bf16_t* dst, const float* w, int K, int N) {  // w[K,N] -> dst[N,K] bf16
    long long n = (long long)K * N;
    k_transpose_bf16<<<dim3((unsigned)((n + 255) / 256), 1, 1), 256, 0, stream>>>(
        dst, w, K, N, N, 0LL, 0LL, 1, 0LL);
  };
  auto gemm = [&](const bf16_t* A, long long aS1, long long aS2, int lda,
                  const bf16_t* Bt, long long bS1, long long bS2, int ldb,
                  float* Cp, long long cS1, long long cS2, int ldc,
                  const float* bias, int M, int N, int K, int batch, int inner,
                  int act, int res) {
    dim3 grd((N + 127) / 128, (M + 63) / 64, batch);
    k_gemm<<<grd, 256, 0, stream>>>(A, aS1, aS2, lda, Bt, bS1, bS2, ldb,
                                    Cp, cS1, cS2, ldc, bias, M, N, K, inner, act, res);
  };
  auto dense = [&](const bf16_t* A, const bf16_t* Wt, float* Cp, const float* bias,
                   int M, int N, int K, int act, int res) {
    gemm(A, 0, 0, K, Wt, 0, 0, K, Cp, 0, 0, N, bias, M, N, K, 1, 1, act, res);
  };
  auto ln = [&](const float* src, LnP p, int rows, int Dn, float* y, bf16_t* ybf) {
    k_layernorm<<<dim3(rows), 256, 0, stream>>>(src, p.g, p.b, y, ybf, Dn);
  };

  // ----- pack weights to bf16 [N,K] -----
  auto wt = [&](const float* w, int K, int N) {
    bf16_t* p = (bf16_t*)alloc((size_t)K * N * 2);
    packT(p, w, K, N);
    return p;
  };
  bf16_t* wt_vproj = wt(video_proj.w, D, D);
  bf16_t* wt_tproj = wt(text_proj.w, D, D);
  bf16_t* wt_tglob = wt(text_global.w, D, D);
  bf16_t* wt_rglob = wt(ref_global.w, D, D);
  struct LayerW { bf16_t *attn_in, *attn_out, *cq, *ck, *cv, *top, *vsp, *g1, *g2, *f1, *f2; };
  LayerW LW[2];
  for (int l = 0; l < 2; ++l) {
    LW[l].attn_in  = wt(L[l].attn_in.w, D, D3);
    LW[l].attn_out = wt(L[l].attn_out.w, D, D);
    LW[l].cq  = wt(L[l].cq.w, D, D);
    LW[l].ck  = wt(L[l].ck.w, D, D);
    LW[l].cv  = wt(L[l].cv.w, D, D);
    LW[l].top = wt(L[l].top.w, D, D);
    LW[l].vsp = wt(L[l].vsp.w, D, C);
    LW[l].g1  = wt(L[l].g1.w, D2, HID);
    LW[l].g2  = wt(L[l].g2.w, HID, 1);
    LW[l].f1  = wt(L[l].f1.w, D, HID);
    LW[l].f2  = wt(L[l].f2.w, HID, D);
  }
  HeadP heads[4] = {h_remaining, h_stop, h_activity, h_viseme};
  int   headN[4] = {1, 1, 1, C};
  bf16_t* wt_hfc1[4]; bf16_t* wt_hfc2[4];
  for (int h = 0; h < 4; ++h) {
    wt_hfc1[h] = wt(heads[h].fc1.w, D, HID);
    wt_hfc2[h] = wt(heads[h].fc2.w, HID, headN[h]);
  }

  // ----- output regions -----
  float* o_hidden = (float*)d_out;
  float* o_logits = o_hidden + (size_t)BS * D;
  float* o_ep     = o_logits + (size_t)BS * T;
  float* o_ent    = o_ep + BS;
  float* o_mono   = o_ent + BS;
  float* o_rem    = o_mono + 1;
  float* o_stop   = o_rem + BS;
  float* o_act    = o_stop + BS;
  float* o_vis    = o_act + BS;
  float* o_cmean  = o_vis + (size_t)BS * C;

  k_setzero<<<1, 256, 0, stream>>>(f_acc, 1);

  // ----- stem -----
  ln(in_video, input_norm, BS, D, nullptr, b_d0);                         // ln(video)
  dense(b_d0, wt_vproj, f_x, video_proj.b, BS, D, D, /*gelu*/1, 0);       // x = gelu(vproj)
  cvt(b_tmin, in_tm, (long long)BT * D);
  dense(b_tmin, wt_tproj, f_tm, text_proj.b, BT, D, D, 0, 0);             // tm
  cvt(b_tm, f_tm, (long long)BT * D);
  cvt(b_ts, in_tsum, (long long)B * D);
  cvt(b_rs, in_rsum, (long long)B * D);
  dense(b_ts, wt_tglob, f_tg, text_global.b, B, D, D, 0, 0);
  dense(b_rs, wt_rglob, f_rg, ref_global.b, B, D, D, 0, 0);
  {
    long long n = (long long)BS * D;
    k_add_rowvec<<<dim3((unsigned)((n + 255) / 256)), 256, 0, stream>>>(f_x, f_tg, S * D, D, n);
    k_add_rowvec<<<dim3((unsigned)((n + 255) / 256)), 256, 0, stream>>>(f_x, f_rg, S * D, D, n);
  }
  ln(f_x, context_norm, BS, D, f_x, nullptr);                             // in-place ok (row-private)

  const float self_scale  = 1.0f / 8.0f;           // 1/sqrt(HD)
  const float cross_scale = 1.0f / 32.0f;          // 1/sqrt(D)
  const float inv2r2 = 1.0f / (2.0f * 4.0f * 4.0f);

  // ----- layers -----
  for (int l = 0; l < 2; ++l) {
    // self-attention
    ln(f_x, L[l].san, BS, D, nullptr, b_d0);
    dense(b_d0, LW[l].attn_in, f_qkv, L[l].attn_in.b, BS, D3, D, 0, 0);
    cvt(b_qkv, f_qkv, (long long)BS * D3);
    // scores[b,h] = q @ k^T   (A=q, Wt=k as [S,HD] rows)
    gemm(b_qkv,     (long long)S * D3, HD, D3,
         b_qkv + D, (long long)S * D3, HD, D3,
         f_sc, (long long)NH * S * S, (long long)S * S, S,
         nullptr, S, S, HD, B * NH, NH, 0, 0);
    k_softmax_self<<<dim3(S, NH, B), 256, 0, stream>>>(f_sc, b_sc, vlens, S, self_scale);
    // v^T per (b,h): [S,HD] -> [HD,S]
    {
      long long n = (long long)S * HD;
      k_transpose_bf16<<<dim3((unsigned)((n + 255) / 256), 1, B * NH), 256, 0, stream>>>(
          b_vt, f_qkv + 2 * D, S, HD, D3, (long long)S * D3, (long long)HD, NH, (long long)HD * S);
    }
    // ctx[b,h] = probs @ v
    gemm(b_sc, (long long)NH * S * S, (long long)S * S, S,
         b_vt, (long long)NH * HD * S, (long long)HD * S, S,
         f_a, (long long)S * D, HD, D,
         nullptr, S, HD, S, B * NH, NH, 0, 0);
    cvt(b_d1, f_a, (long long)BS * D);
    dense(b_d1, LW[l].attn_out, f_x, L[l].attn_out.b, BS, D, D, 0, /*res*/1);

    // cursor cross-attention
    cvt(b_d0, f_x, (long long)BS * D);                    // x bf16
    dense(b_d0, LW[l].cq, f_a, L[l].cq.b, BS, D, D, 0, 0);        // q2
    dense(b_tm, LW[l].ck, f_k2, L[l].ck.b, BT, D, D, 0, 0);       // k2
    dense(b_tm, LW[l].cv, f_v2, L[l].cv.b, BT, D, D, 0, 0);       // v2
    cvt(b_d1, f_a, (long long)BS * D);
    cvt(b_k2, f_k2, (long long)BT * D);
    gemm(b_d1, (long long)S * D, 0, D,
         b_k2, (long long)T * D, 0, D,
         f_lg, (long long)S * T, 0, T,
         nullptr, S, T, D, B, 1, 0, 0);                   // logits = q2 @ k2^T
    dense(b_d0, LW[l].vsp, f_vlog, L[l].vsp.b, BS, C, D, 0, 0);   // vlog
    k_cross<<<dim3(S, B), 256, 0, stream>>>(
        f_lg, f_vlog, in_tok, in_vtab, tlens, f_ep, f_lw, b_lw, f_ent, f_acc,
        S, T, C, V, cross_scale, 0.5f, inv2r2, (l == 1) ? 1 : 0);
    {
      long long n = (long long)T * D;
      k_transpose_bf16<<<dim3((unsigned)((n + 255) / 256), 1, B), 256, 0, stream>>>(
          b_v2t, f_v2, T, D, D, (long long)T * D, 0LL, 1, (long long)D * T);
    }
    gemm(b_lw, (long long)S * T, 0, T,
         b_v2t, (long long)D * T, 0, T,
         f_a, (long long)S * D, 0, D,
         nullptr, S, D, T, B, 1, 0, 0);                   // lw @ v2
    cvt(b_d1, f_a, (long long)BS * D);
    dense(b_d1, LW[l].top, f_tctx, L[l].top.b, BS, D, D, 0, 0);   // tctx

    // gated merge
    {
      long long n = (long long)BS * D2;
      k_concat2<<<dim3((unsigned)((n + 255) / 256)), 256, 0, stream>>>(f_gi, f_x, f_tctx, D2, D, n);
    }
    ln(f_gi, L[l].gn, BS, D2, nullptr, b_gi);
    dense(b_gi, LW[l].g1, f_big, L[l].g1.b, BS, HID, D2, /*gelu*/1, 0);
    cvt(b_big, f_big, (long long)BS * HID);
    dense(b_big, LW[l].g2, f_gate, L[l].g2.b, BS, 1, HID, /*sigmoid*/2, 0);
    {
      long long n = (long long)BS * D;
      k_gate_merge<<<dim3((unsigned)((n + 255) / 256)), 256, 0, stream>>>(f_x, f_tctx, f_gate, D, n);
    }

    // FFN
    ln(f_x, L[l].fn, BS, D, nullptr, b_d0);
    dense(b_d0, LW[l].f1, f_big, L[l].f1.b, BS, HID, D, /*gelu*/1, 0);
    cvt(b_big, f_big, (long long)BS * HID);
    dense(b_big, LW[l].f2, f_x, L[l].f2.b, BS, D, HID, 0, /*res*/1);
  }

  // ----- outputs -----
  ln(f_x, output_norm, BS, D, o_hidden, nullptr);
  float* headOut[4] = {o_rem, o_stop, o_act, o_vis};
  for (int h = 0; h < 4; ++h) {
    ln(o_hidden, heads[h].ln, BS, D, nullptr, b_d1);
    dense(b_d1, wt_hfc1[h], f_big, heads[h].fc1.b, BS, HID, D, /*gelu*/1, 0);
    cvt(b_big, f_big, (long long)BS * HID);
    dense(b_big, wt_hfc2[h], headOut[h], heads[h].fc2.b, BS, headN[h], HID, 0, 0);
  }
  {
    long long n = (long long)BS * T;
    k_copy<<<dim3((unsigned)((n + 255) / 256)), 256, 0, stream>>>(o_logits, f_lg, n);
    k_copy<<<dim3((unsigned)((BS + 255) / 256)), 256, 0, stream>>>(o_ep, f_ep, BS);
    k_copy<<<dim3((unsigned)((BS + 255) / 256)), 256, 0, stream>>>(o_ent, f_ent, BS);
  }
  k_mono<<<1, 256, 0, stream>>>(f_ep, vlens, o_mono, B, S);
  k_scale_from<<<1, 1, 0, stream>>>(o_cmean, f_acc, 1.0f / (float)((long long)BS * T));
  (void)in_sizes; (void)n_in; (void)out_size; (void)ws_size;
}
